// SelfAttention_27178553049843
// MI455X (gfx1250) — compile-verified, matrix operand order checked by reading, not run on hardware
//
#include <hip/hip_runtime.h>

#define BDIM 8
#define TDIM 2048
#define EDIM 1024
#define HDIM 64

typedef __attribute__((ext_vector_type(16))) __bf16 v16bf;
typedef __attribute__((ext_vector_type(8)))  float  v8f;

union BF16x16 { v16bf v; unsigned u[8]; };

__device__ __forceinline__ unsigned short bf16bits(float f) {
  union { __bf16 h; unsigned short s; } c; c.h = (__bf16)f; return c.s;
}
__device__ __forceinline__ unsigned packbf16(float a, float b) {
  union { __bf16 h[2]; unsigned u; } p; p.h[0] = (__bf16)a; p.h[1] = (__bf16)b; return p.u;
}
__device__ __forceinline__ float redmax16(float v) {
#pragma unroll
  for (int m = 1; m < 16; m <<= 1) v = fmaxf(v, __shfl_xor(v, m, 32));
  return v;
}
__device__ __forceinline__ float redsum16(float v) {
#pragma unroll
  for (int m = 1; m < 16; m <<= 1) v += __shfl_xor(v, m, 32);
  return v;
}
__device__ __forceinline__ void async_b128(unsigned lds, unsigned long long ga) {
  asm volatile("global_load_async_to_lds_b128 %0, %1, off" :: "v"(lds), "v"(ga) : "memory");
}

// ---------------------------------------------------------------------------
// Kernel 1: fused QKV projection, fp32 -> bf16, q pre-scaled by 1/sqrt(H).
// ---------------------------------------------------------------------------
__device__ __forceinline__ void load_wfrag(BF16x16& Bf, const unsigned short* wl,
                                           int j, int m, int half8) {
  const unsigned short* base = wl + ((j >> 2) * 64 + (j & 3) * 16 + m) * 34;
#pragma unroll
  for (int r = 0; r < 8; r++) {
    int kk = ((r < 4) ? 0 : 16) + half8 + (r & 3) * 2;
    Bf.u[r] = *(const unsigned*)(base + kk);
  }
}

__global__ __launch_bounds__(256) void qkv_proj_kernel(
    const float* __restrict__ x,
    const float* __restrict__ Wk, const float* __restrict__ bk,
    const float* __restrict__ Wq, const float* __restrict__ bq,
    const float* __restrict__ Wv, const float* __restrict__ bv,
    unsigned short* __restrict__ qws,   // [B*T][H] bf16 (scaled by 0.125)
    unsigned short* __restrict__ kws,   // [B*T][H] bf16
    unsigned short* __restrict__ vws)   // [B][H][T] bf16 (transposed)
{
  __shared__ unsigned short wlds[2][3 * 64 * 34];  // [buf][proj][col][kk], 34-elem stride

  const int tid   = threadIdx.x;
  const int wave  = tid >> 5;
  const int lane  = tid & 31;
  const int m     = lane & 15;
  const int half8 = (lane < 16) ? 0 : 8;
  const int rowblock = blockIdx.x * 128;

  const float* Ws[3] = {Wq, Wk, Wv};
  const int skk  = tid >> 3;        // 0..31
  const int scol = (tid & 7) * 8;   // 0..56

  v8f acc[3][4];
#pragma unroll
  for (int p = 0; p < 3; p++)
#pragma unroll
    for (int nt = 0; nt < 4; nt++)
#pragma unroll
      for (int i = 0; i < 8; i++) acc[p][nt][i] = 0.0f;

  // stage k-step 0 into buffer 0
#pragma unroll
  for (int p = 0; p < 3; p++) {
    const float* src = Ws[p] + (size_t)skk * HDIM + scol;
#pragma unroll
    for (int i = 0; i < 8; i++)
      wlds[0][(p * 64 + scol + i) * 34 + skk] = bf16bits(src[i]);
  }
  __syncthreads();

  const float* xr = x + (size_t)(rowblock + wave * 16 + m) * EDIM;

  for (int ks = 0; ks < 32; ks++) {
    const int cur = ks & 1;
    const unsigned short* wl = &wlds[cur][0];

    // A fragment straight from global (lanes l and l+16 split the 128B row line)
    BF16x16 A;
    {
      const float* xc = xr + ks * 32;
      float4 f0 = *(const float4*)(xc + half8);
      float4 f1 = *(const float4*)(xc + half8 + 4);
      float4 f2 = *(const float4*)(xc + 16 + half8);
      float4 f3 = *(const float4*)(xc + 16 + half8 + 4);
      A.u[0] = packbf16(f0.x, f0.y); A.u[1] = packbf16(f0.z, f0.w);
      A.u[2] = packbf16(f1.x, f1.y); A.u[3] = packbf16(f1.z, f1.w);
      A.u[4] = packbf16(f2.x, f2.y); A.u[5] = packbf16(f2.z, f2.w);
      A.u[6] = packbf16(f3.x, f3.y); A.u[7] = packbf16(f3.z, f3.w);
    }

    // 12 WMMA jobs (p*4+nt), B-fragments pipelined through a register double buffer
    BF16x16 Bf[2];
    load_wfrag(Bf[0], wl, 0, m, half8);
#pragma unroll
    for (int j = 0; j < 12; j++) {
      if (j + 1 < 12) load_wfrag(Bf[(j + 1) & 1], wl, j + 1, m, half8);
      acc[j >> 2][j & 3] = __builtin_amdgcn_wmma_f32_16x16x32_bf16(
          false, A.v, false, Bf[j & 1].v, (short)0, acc[j >> 2][j & 3], false, false);
    }

    // stage next W chunk AFTER compute (keeps ds waits precise during WMMAs)
    if (ks + 1 < 32) {
#pragma unroll
      for (int p = 0; p < 3; p++) {
        const float* src = Ws[p] + (size_t)((ks + 1) * 32 + skk) * HDIM + scol;
#pragma unroll
        for (int i = 0; i < 8; i++)
          wlds[cur ^ 1][(p * 64 + scol + i) * 34 + skk] = bf16bits(src[i]);
      }
    }
    __syncthreads();
  }

  const float* biases[3] = {bq, bk, bv};
#pragma unroll
  for (int p = 0; p < 3; p++) {
#pragma unroll
    for (int nt = 0; nt < 4; nt++) {
      int col = nt * 16 + m;
      float bias = biases[p][col];
#pragma unroll
      for (int r = 0; r < 8; r++) {
        int trow = rowblock + wave * 16 + r + half8;
        float val = acc[p][nt][r] + bias;
        if (p == 0) {
          qws[(size_t)trow * HDIM + col] = bf16bits(val * 0.125f);
        } else if (p == 1) {
          kws[(size_t)trow * HDIM + col] = bf16bits(val);
        } else {
          int b = trow >> 11, t = trow & (TDIM - 1);
          vws[((size_t)b * HDIM + col) * TDIM + t] = bf16bits(val);
        }
      }
    }
  }
}

// ---------------------------------------------------------------------------
// Kernel 2: causal flash attention. Block = 128 query rows (8 waves x 16),
// key tiles of 64, K/V async-to-LDS double-buffered (1 barrier / tile).
// ---------------------------------------------------------------------------
__device__ __forceinline__ void load_kvfrag(BF16x16& F0, BF16x16& F1,
                                            const unsigned char* base, int col, int half8) {
  const unsigned char* rowp = base + col * 128;
#pragma unroll
  for (int r = 0; r < 8; r++) {
    int kk = ((r < 4) ? 0 : 16) + half8 + (r & 3) * 2;
    int by0 = kk * 2, by1 = (kk + 32) * 2;
    F0.u[r] = *(const unsigned*)(rowp + ((((by0 >> 4) ^ (col & 7)) << 4) | (by0 & 15)));
    F1.u[r] = *(const unsigned*)(rowp + ((((by1 >> 4) ^ (col & 7)) << 4) | (by1 & 15)));
  }
}

__global__ __launch_bounds__(256) void flash_attn_kernel(
    const unsigned short* __restrict__ qws,
    const unsigned short* __restrict__ kws,
    const unsigned short* __restrict__ vws,   // [B][H][T]
    float* __restrict__ out)
{
  __shared__ __align__(128) unsigned char klds[2][64 * 128]; // [key][h] bf16, 16B-chunk swizzle
  __shared__ __align__(128) unsigned char vlds[2][64 * 128]; // [h][key] bf16, 16B-chunk swizzle
  __shared__ unsigned short plds[8 * 16 * 66];               // per-wave P staging, padded

  const int tid   = threadIdx.x;
  const int wave  = tid >> 5;
  const int lane  = tid & 31;
  const int m     = lane & 15;
  const int half8 = (lane < 16) ? 0 : 8;
  const int b     = blockIdx.y;
  const int qtile = blockIdx.x;
  const int qwbase = qtile * 128 + wave * 16;
  const int qwmax  = qwbase + 15;

  // Preload Q A-fragments (two K=32 steps covering H=64)
  BF16x16 QA[2];
  {
    const unsigned short* qrow = qws + ((size_t)b * TDIM + qwbase + m) * HDIM;
#pragma unroll
    for (int ks = 0; ks < 2; ks++)
#pragma unroll
      for (int r = 0; r < 8; r++) {
        int kk = ks * 32 + ((r < 4) ? 0 : 16) + half8 + (r & 3) * 2;
        QA[ks].u[r] = *(const unsigned*)(qrow + kk);
      }
  }

  v8f O[4];
  float mrow[8], lrow[8];
#pragma unroll
  for (int nt = 0; nt < 4; nt++)
#pragma unroll
    for (int i = 0; i < 8; i++) O[nt][i] = 0.0f;
#pragma unroll
  for (int r = 0; r < 8; r++) { mrow[r] = -1e30f; lrow[r] = 0.0f; }

  const unsigned kbase = (unsigned)(size_t)(void*)&klds[0][0];
  const unsigned vbase = (unsigned)(size_t)(void*)&vlds[0][0];
  unsigned short* pw = plds + wave * (16 * 66);

  const unsigned long long kgb = (unsigned long long)(size_t)(kws + (size_t)b * TDIM * HDIM);
  const unsigned long long vgb = (unsigned long long)(size_t)(vws + (size_t)b * HDIM * TDIM);

  const int ntiles = (qtile + 1) * 2;   // causal: s0 < qbase + 128

  // prologue: load tile 0 into buffer 0
#pragma unroll
  for (int i = 0; i < 2; i++) {
    int idx = tid + 256 * i, row = idx >> 3, part = idx & 7;
    async_b128(kbase + row * 128 + ((part ^ (row & 7)) << 4),
               kgb + (unsigned)(row * 128 + part * 16));
    async_b128(vbase + row * 128 + ((part ^ (row & 7)) << 4),
               vgb + (unsigned)(row * TDIM * 2 + part * 16));
  }
  asm volatile("s_wait_asynccnt 0" ::: "memory");
  __syncthreads();

  for (int st = 0; st < ntiles; st++) {
    const int s0 = st * 64;
    const unsigned koff = (unsigned)((st & 1) * (64 * 128));
    // issue async loads for next tile into the other buffer
    if (st + 1 < ntiles) {
      const unsigned noff = (unsigned)(((st + 1) & 1) * (64 * 128));
#pragma unroll
      for (int i = 0; i < 2; i++) {
        int idx = tid + 256 * i, row = idx >> 3, part = idx & 7;
        unsigned swz = (unsigned)(row * 128 + ((part ^ (row & 7)) << 4));
        async_b128(kbase + noff + swz,
                   kgb + (unsigned)(((s0 + 64 + row) * HDIM + part * 8) * 2));
        async_b128(vbase + noff + swz,
                   vgb + (unsigned)((row * TDIM + s0 + 64 + part * 8) * 2));
      }
    }

    if (s0 <= qwmax) {
      const unsigned char* kb = &klds[0][0] + koff;
      const unsigned char* vb = &vlds[0][0] + koff;

      // S = Q * K^T : 4 N-tiles, K-fragments pipelined through register double buffer
      v8f S[4];
      {
        BF16x16 KB[2][2];
        load_kvfrag(KB[0][0], KB[0][1], kb, m, half8);
#pragma unroll
        for (int nt = 0; nt < 4; nt++) {
          if (nt + 1 < 4)
            load_kvfrag(KB[(nt + 1) & 1][0], KB[(nt + 1) & 1][1], kb, (nt + 1) * 16 + m, half8);
          v8f s;
#pragma unroll
          for (int i = 0; i < 8; i++) s[i] = 0.0f;
          s = __builtin_amdgcn_wmma_f32_16x16x32_bf16(
              false, QA[0].v, false, KB[nt & 1][0].v, (short)0, s, false, false);
          s = __builtin_amdgcn_wmma_f32_16x16x32_bf16(
              false, QA[1].v, false, KB[nt & 1][1].v, (short)0, s, false, false);
          S[nt] = s;
        }
      }

      // causal mask + online softmax
#pragma unroll
      for (int r = 0; r < 8; r++) {
        int rowg = qwbase + r + half8;
        float sv[4], pv[4];
        float rm = mrow[r];
#pragma unroll
        for (int nt = 0; nt < 4; nt++) {
          sv[nt] = ((s0 + nt * 16 + m) <= rowg) ? S[nt][r] : -1e30f;
          rm = fmaxf(rm, sv[nt]);
        }
        rm = redmax16(rm);
        float alpha = __expf(mrow[r] - rm);
        float rs = 0.0f;
#pragma unroll
        for (int nt = 0; nt < 4; nt++) { pv[nt] = __expf(sv[nt] - rm); rs += pv[nt]; }
        rs = redsum16(rs);
        mrow[r] = rm;
        lrow[r] = lrow[r] * alpha + rs;
        O[0][r] *= alpha; O[1][r] *= alpha; O[2][r] *= alpha; O[3][r] *= alpha;
#pragma unroll
        for (int nt = 0; nt < 4; nt++)
          pw[(r + half8) * 66 + nt * 16 + m] = bf16bits(pv[nt]);
      }
      asm volatile("" ::: "memory");   // LDS in-order within a wave

      // P as two A-fragments (rows = queries, K = 64 keys)
      BF16x16 PA[2];
#pragma unroll
      for (int ks = 0; ks < 2; ks++)
#pragma unroll
        for (int r = 0; r < 8; r++) {
          int kk = ks * 32 + ((r < 4) ? 0 : 16) + half8 + (r & 3) * 2;
          PA[ks].u[r] = *(const unsigned*)(pw + m * 66 + kk);
        }

      // O += P * V : 4 N-tiles, V-fragments pipelined through register double buffer
      {
        BF16x16 VB[2][2];
        load_kvfrag(VB[0][0], VB[0][1], vb, m, half8);
#pragma unroll
        for (int nt2 = 0; nt2 < 4; nt2++) {
          if (nt2 + 1 < 4)
            load_kvfrag(VB[(nt2 + 1) & 1][0], VB[(nt2 + 1) & 1][1], vb, (nt2 + 1) * 16 + m, half8);
          O[nt2] = __builtin_amdgcn_wmma_f32_16x16x32_bf16(
              false, PA[0].v, false, VB[nt2 & 1][0].v, (short)0, O[nt2], false, false);
          O[nt2] = __builtin_amdgcn_wmma_f32_16x16x32_bf16(
              false, PA[1].v, false, VB[nt2 & 1][1].v, (short)0, O[nt2], false, false);
        }
      }
    }

    asm volatile("s_wait_asynccnt 0" ::: "memory");  // own next-tile loads landed
    __syncthreads();                                 // everyone done + all loads visible
  }

  // normalize + store fp32
#pragma unroll
  for (int nt2 = 0; nt2 < 4; nt2++) {
    int hc = nt2 * 16 + m;
#pragma unroll
    for (int r = 0; r < 8; r++) {
      int rowg = qwbase + r + half8;
      out[((size_t)b * TDIM + rowg) * HDIM + hc] = O[nt2][r] / lrow[r];
    }
  }
}

extern "C" void kernel_launch(void* const* d_in, const int* in_sizes, int n_in,
                              void* d_out, int out_size, void* d_ws, size_t ws_size,
                              hipStream_t stream) {
  const float* x  = (const float*)d_in[0];
  const float* Wk = (const float*)d_in[1];
  const float* bk = (const float*)d_in[2];
  const float* Wq = (const float*)d_in[3];
  const float* bq = (const float*)d_in[4];
  const float* Wv = (const float*)d_in[5];
  const float* bv = (const float*)d_in[6];
  float* out = (float*)d_out;

  const size_t nqkv = (size_t)BDIM * TDIM * HDIM;   // 1,048,576 elems (2 MB bf16 each)
  unsigned short* qws = (unsigned short*)d_ws;
  unsigned short* kws = qws + nqkv;
  unsigned short* vws = kws + nqkv;

  qkv_proj_kernel<<<(BDIM * TDIM) / 128, 256, 0, stream>>>(
      x, Wk, bk, Wq, bq, Wv, bv, qws, kws, vws);
  flash_attn_kernel<<<dim3(TDIM / 128, BDIM), 256, 0, stream>>>(
      qws, kws, vws, out);
}
